// FastRegionCNN_71296457113876
// MI455X (gfx1250) — matrix operand order repaired and checked by reading, not compiled, not run on hardware
//
#include <hip/hip_runtime.h>
#include <hip/hip_bf16.h>
#include <math.h>

// ---- CDNA5 WMMA vector types (wave32) ----
typedef __attribute__((ext_vector_type(16))) __bf16 v16bf;
typedef __attribute__((ext_vector_type(8)))  __bf16 bf16x8;
typedef __attribute__((ext_vector_type(2)))  __bf16 bf16x2;
typedef __attribute__((ext_vector_type(8)))  float  v8f;
typedef __attribute__((ext_vector_type(4)))  float  f32x4;

#define RP    7          // roi_output_size
#define NCLS  21
#define KBOX  128
#define CCH   512
#define HH    32
#define WW    32
#define D1    (CCH*RP*RP)   // 25088
#define NH    4096
#define SPLITK 4

// ---------------------------------------------------------------------------
// Kernel 1: ROI max pool -> bf16 activations [KBOX, D1] row-major
// ---------------------------------------------------------------------------
__global__ __launch_bounds__(256)
void roi_pool_kernel(const float* __restrict__ feat,   // [4,512,32,32]
                     const float* __restrict__ boxes,  // [128,5]
                     __bf16* __restrict__ xout)        // [128, 25088] bf16
{
    const int box = blockIdx.x;
    const float* bx = boxes + box * 5;
    const int   b  = (int)bx[0];
    const float x1 = rintf(bx[1]), y1 = rintf(bx[2]);   // RNE matches jnp.round
    const float x2 = rintf(bx[3]), y2 = rintf(bx[4]);
    const float rw = fmaxf(x2 - x1 + 1.f, 1.f);
    const float rh = fmaxf(y2 - y1 + 1.f, 1.f);
    const float* fb = feat + (size_t)b * CCH * HH * WW;

    for (int idx = threadIdx.x; idx < D1; idx += blockDim.x) {
        const int c  = idx / (RP * RP);
        const int r  = idx % (RP * RP);
        const int ph = r / RP, pw = r % RP;
        const int hs = (int)fminf(fmaxf(floorf(ph * rh / (float)RP) + y1, 0.f), (float)HH);
        const int he = (int)fminf(fmaxf(ceilf((ph + 1) * rh / (float)RP) + y1, 0.f), (float)HH);
        const int wls = (int)fminf(fmaxf(floorf(pw * rw / (float)RP) + x1, 0.f), (float)WW);
        const int wle = (int)fminf(fmaxf(ceilf((pw + 1) * rw / (float)RP) + x1, 0.f), (float)WW);
        float m = -INFINITY;
        const float* fc = fb + (size_t)c * HH * WW;
        for (int h = hs; h < he; ++h)
            for (int w = wls; w < wle; ++w)
                m = fmaxf(m, fc[h * WW + w]);
        if (he <= hs || wle <= wls) m = 0.f;   // empty bin -> 0 (torchvision)
        xout[(size_t)box * D1 + idx] = (__bf16)m;
    }
}

// ---------------------------------------------------------------------------
// Kernel 2: split-K GEMM partial:
//   part[kc, 128, N](f32) = A[128, kc-chunk](bf16) @ Bw[kc-chunk, N](f32->bf16)
//
// grid = (N/64, SPLITK) => 256 WGs of 8 wave32 -> enough MLP to pull 23.3TB/s.
// Per WG: all 128 rows x 64 cols of its K-chunk; every weight element is
// streamed from HBM exactly once across the whole grid.
// Weight tile (32K x 64N) staged cooperatively via coalesced b128 f32 loads +
// v_cvt_pk_bf16_f32, stored TRANSPOSED in LDS (n-major, k-contiguous, 80B row
// stride: 16B aligned, bank-spread); B fragments = 2x ds_load_b128. Double-
// buffered so next tile's global loads issue before the WMMA quartet.
// ---------------------------------------------------------------------------
template<int KC, int N>
__global__ __launch_bounds__(256)
void gemm_bf16_wmma(const __bf16* __restrict__ A,    // [128, SPLITK*KC] bf16
                    const float*  __restrict__ Bw,   // [SPLITK*KC, N] f32
                    float* __restrict__ part)        // [SPLITK, 128, N] f32
{
    constexpr int LSTR = 40;            // LDS row stride in bf16 (80B)
    constexpr int LBUF = 64 * LSTR;     // one buffer: 64 n-rows
    constexpr int KD   = SPLITK * KC;
    __shared__ __bf16 smem[2 * LBUF];

    const int tid    = threadIdx.x;
    const int lane   = tid & 31;
    const int wave   = tid >> 5;        // 0..7 -> M subtile
    const int mbase  = wave * 16;
    const int nbase  = blockIdx.x * 64;
    const int kb0    = blockIdx.y * KC; // this WG's K-chunk
    const int lane16 = lane & 15;
    const int hi     = lane >> 4;       // lane group 0/1

    // staging role: this thread covers K rows (kp, kp+1) x 4 consecutive n
    const int kp = (tid >> 4) * 2;      // even k in [0,31]
    const int n0 = (tid & 15) * 4;

    v8f acc[4] = {v8f{}, v8f{}, v8f{}, v8f{}};

    // A-fragment base (16x32 bf16 tile, ISA layout):
    //   lanes 0-15 : elems 0..7 = K+0..7,  elems 8..15 = K+16..23
    //   lanes16-31 : elems 0..7 = K+8..15, elems 8..15 = K+24..31
    const __bf16* arow = A + (size_t)(mbase + lane16) * KD + kb0 + hi * 8;
    const float*  bchunk = Bw + (size_t)kb0 * N;

    // ---- prologue: stage first K-tile into buffer 0 ----
    {
        const float* g = bchunk + (size_t)kp * N + nbase + n0;
        f32x4 w0 = *(const f32x4*)g;
        f32x4 w1 = *(const f32x4*)(g + N);
        #pragma unroll
        for (int i = 0; i < 4; ++i) {
            bf16x2 p; p[0] = (__bf16)w0[i]; p[1] = (__bf16)w1[i];
            *(bf16x2*)&smem[(n0 + i) * LSTR + kp] = p;
        }
    }
    __syncthreads();

    int buf = 0;
    for (int kb = 0; kb < KC; kb += 32) {
        const bool more = (kb + 32) < KC;

        // issue next tile's global loads early (latency hides under WMMAs)
        f32x4 w0, w1;
        if (more) {
            const float* g = bchunk + (size_t)(kb + 32 + kp) * N + nbase + n0;
            w0 = *(const f32x4*)g;
            w1 = *(const f32x4*)(g + N);
            if (kb + 64 < KC)  // gfx1250 global_prefetch_b8, tile after next
                __builtin_prefetch((const void*)(g + (size_t)32 * N), 0, 0);
        }

        // A fragment (global; 6.4MB activations are L2-resident)
        v16bf afrag;
        ((bf16x8*)&afrag)[0] = *(const bf16x8*)(arow + kb);
        ((bf16x8*)&afrag)[1] = *(const bf16x8*)(arow + kb + 16);

        // B fragments: lane -> n = t*16+lane16, elems = K hi*8+0..7, 16+hi*8+0..7
        // (load all four first so the ds_load_b128s issue as one clause)
        const __bf16* bbase = &smem[buf * LBUF + hi * 8];
        v16bf bfrag[4];
        #pragma unroll
        for (int t = 0; t < 4; ++t) {
            const __bf16* bsrc = bbase + (t * 16 + lane16) * LSTR;
            ((bf16x8*)&bfrag[t])[0] = *(const bf16x8*)(bsrc);
            ((bf16x8*)&bfrag[t])[1] = *(const bf16x8*)(bsrc + 16);
        }
        #pragma unroll
        for (int t = 0; t < 4; ++t)
            acc[t] = __builtin_amdgcn_wmma_f32_16x16x32_bf16(
                         false, afrag, false, bfrag[t],
                         (short)0, acc[t], false, false);

        // convert + store next tile (transposed) into the other buffer
        if (more) {
            __bf16* dst = &smem[(buf ^ 1) * LBUF];
            #pragma unroll
            for (int i = 0; i < 4; ++i) {
                bf16x2 p; p[0] = (__bf16)w0[i]; p[1] = (__bf16)w1[i];
                *(bf16x2*)&dst[(n0 + i) * LSTR + kp] = p;
            }
        }
        __syncthreads();   // writes visible before next reads; reads done before overwrite
        buf ^= 1;
    }

    // Store f32 partial block. C/D layout: VGPR r -> row mbase + r + 8*hi,
    // col = nbase + t*16 + lane16.
    float* pout = part + (size_t)blockIdx.y * KBOX * N;
    #pragma unroll
    for (int t = 0; t < 4; ++t) {
        const int col = nbase + t * 16 + lane16;
        #pragma unroll
        for (int r = 0; r < 8; ++r) {
            const int row = mbase + r + hi * 8;
            pout[(size_t)row * N + col] = acc[t][r];
        }
    }
}

// ---------------------------------------------------------------------------
// Kernel 2b: reduce split-K partials, + bias, ReLU, -> bf16
// ---------------------------------------------------------------------------
template<int N>
__global__ __launch_bounds__(256)
void reduce_bias_relu(const float* __restrict__ part, // [SPLITK,128,N]
                      const float* __restrict__ bias, // [N]
                      __bf16* __restrict__ Cb)        // [128,N] bf16
{
    const int idx = blockIdx.x * 256 + threadIdx.x;   // over 128*N
    float s = 0.f;
    #pragma unroll
    for (int p = 0; p < SPLITK; ++p)
        s += part[(size_t)p * KBOX * N + idx];
    s += bias[idx & (N - 1)];
    s = fmaxf(s, 0.f);
    Cb[idx] = (__bf16)s;
}

// ---------------------------------------------------------------------------
// Kernel 3: classification softmax + bbox regression heads (tiny)
// ---------------------------------------------------------------------------
__global__ __launch_bounds__(128)
void heads_kernel(const __bf16* __restrict__ h2,  // [128,4096] bf16
                  const float* __restrict__ Wc, const float* __restrict__ bc,
                  const float* __restrict__ Wr, const float* __restrict__ br,
                  float* __restrict__ probs,      // [128,21]
                  float* __restrict__ regs)       // [128,84]
{
    __shared__ float hrow[NH];
    __shared__ float logits[NCLS];
    const int m = blockIdx.x;
    for (int k = threadIdx.x; k < NH; k += blockDim.x)
        hrow[k] = (float)h2[(size_t)m * NH + k];
    __syncthreads();

    const int j = threadIdx.x;
    if (j < NCLS) {
        float s = bc[j];
        for (int k = 0; k < NH; ++k) s += hrow[k] * Wc[(size_t)k * NCLS + j];
        logits[j] = s;
    } else if (j < NCLS + 4 * NCLS) {
        const int j2 = j - NCLS;
        float s = br[j2];
        for (int k = 0; k < NH; ++k) s += hrow[k] * Wr[(size_t)k * (4 * NCLS) + j2];
        regs[(size_t)m * (4 * NCLS) + j2] = s;
    }
    __syncthreads();

    if (j < NCLS) {
        float mx = -INFINITY;
        for (int t = 0; t < NCLS; ++t) mx = fmaxf(mx, logits[t]);
        float den = 0.f;
        for (int t = 0; t < NCLS; ++t) den += expf(logits[t] - mx);
        probs[(size_t)m * NCLS + j] = expf(logits[j] - mx) / den;
    }
}

// ---------------------------------------------------------------------------
extern "C" void kernel_launch(void* const* d_in, const int* in_sizes, int n_in,
                              void* d_out, int out_size, void* d_ws, size_t ws_size,
                              hipStream_t stream)
{
    const float* feat  = (const float*)d_in[0];
    const float* boxes = (const float*)d_in[1];
    const float* W1    = (const float*)d_in[2];
    const float* b1    = (const float*)d_in[3];
    const float* W2    = (const float*)d_in[4];
    const float* b2    = (const float*)d_in[5];
    const float* Wc    = (const float*)d_in[6];
    const float* bc    = (const float*)d_in[7];
    const float* Wr    = (const float*)d_in[8];
    const float* br    = (const float*)d_in[9];

    char* wsb = (char*)d_ws;
    __bf16* xbf = (__bf16*)(wsb);                                  // 6.42 MB
    size_t off = (size_t)KBOX * D1 * 2;
    __bf16* h1  = (__bf16*)(wsb + off);  off += (size_t)KBOX * NH * 2;   // 1 MB
    __bf16* h2  = (__bf16*)(wsb + off);  off += (size_t)KBOX * NH * 2;   // 1 MB
    float*  part = (float*)(wsb + off);  // SPLITK*128*4096*4 = 8 MB (reused)

    float* probs = (float*)d_out;              // [128,21]
    float* regs  = probs + KBOX * NCLS;        // [128,84]

    roi_pool_kernel<<<KBOX, 256, 0, stream>>>(feat, boxes, xbf);

    // FC1: K = 25088 = 4 chunks x 6272 (196 K-steps each)
    gemm_bf16_wmma<D1 / SPLITK, NH>
        <<<dim3(NH / 64, SPLITK), 256, 0, stream>>>(xbf, W1, part);
    reduce_bias_relu<NH><<<(KBOX * NH) / 256, 256, 0, stream>>>(part, b1, h1);

    // FC2: K = 4096 = 4 chunks x 1024 (32 K-steps each)
    gemm_bf16_wmma<NH / SPLITK, NH>
        <<<dim3(NH / 64, SPLITK), 256, 0, stream>>>(h1, W2, part);
    reduce_bias_relu<NH><<<(KBOX * NH) / 256, 256, 0, stream>>>(part, b2, h2);

    heads_kernel<<<KBOX, 128, 0, stream>>>(h2, Wc, bc, Wr, br, probs, regs);
}
